// GraphNodes_70179765616691
// MI455X (gfx1250) — matrix operand order repaired and checked by reading, not compiled, not run on hardware
//
#include <hip/hip_runtime.h>
#include <hip/hip_bf16.h>

// ---------------------------------------------------------------------------
// Problem constants (from reference): x [4096,128] f32, W [128,128], b [128]
// h = leaky_relu( A @ (diag(1/Alsum) @ (A.T @ (x@W.T))) + b ),  A = softmax(exp(-dist))
// ---------------------------------------------------------------------------
#define BS   4096
#define FIN  128
#define NEG_SLOPE 0.2f

typedef __bf16 bf16_t;
typedef bf16_t v4bf  __attribute__((ext_vector_type(4)));
typedef bf16_t v8bf  __attribute__((ext_vector_type(8)));
typedef bf16_t v16bf __attribute__((ext_vector_type(16)));
typedef float  v8f   __attribute__((ext_vector_type(8)));

// ---- WMMA helper: D = A(16x32 bf16) x B(32x16 bf16) + C(16x16 f32) ---------
static __device__ __forceinline__ v8f wmma_bf16(v16bf a, v16bf b, v8f c) {
    return __builtin_amdgcn_wmma_f32_16x16x32_bf16(
        /*neg_a=*/false, a, /*neg_b=*/false, b,
        /*c_mod=*/(short)0, c, /*reuse_a=*/false, /*reuse_b=*/false);
}

// A-matrix fragment (16x32, 16-bit, row-major memory, row stride = ldk elems):
// lane L holds M=L%16, contiguous K chunk (L<16: K=k0..k0+15, L>=16: K=k0+16..k0+31).
// Caller passes p = rowptr + k0 + half*16.
static __device__ __forceinline__ v16bf ld_afrag(const bf16_t* p) {
    return *(const v16bf*)(p);
}
// B-matrix fragment (32x16): lane L holds column N=L%16, contiguous K chunk
// (L<16: K=k0..k0+15, L>=16: K=k0+16..k0+31). Caller passes K-contiguous ptr.
static __device__ __forceinline__ v16bf ld_bfrag(const bf16_t* p) {
    return *(const v16bf*)(p);
}

// ---------------------------------------------------------------------------
// K0: xb = bf16(x); sq[i] = sum_k x[i,k]^2.   grid BS, block 32 (1 wave/row)
// ---------------------------------------------------------------------------
__global__ void k_prep(const float* __restrict__ x, bf16_t* __restrict__ xb,
                       float* __restrict__ sq) {
    const int row  = blockIdx.x;
    const int lane = threadIdx.x;
    const float4 v = ((const float4*)(x + (size_t)row * FIN))[lane];
    v4bf o;
    o[0] = (bf16_t)v.x; o[1] = (bf16_t)v.y; o[2] = (bf16_t)v.z; o[3] = (bf16_t)v.w;
    *(v4bf*)(xb + (size_t)row * FIN + lane * 4) = o;
    float ss = v.x*v.x + v.y*v.y + v.z*v.z + v.w*v.w;
    #pragma unroll
    for (int m = 16; m >= 1; m >>= 1) ss += __shfl_xor(ss, m, 32);
    if (lane == 0) sq[row] = ss;
}

// ---------------------------------------------------------------------------
// K1: s = x @ W.T  (f32 accumulate), stored transposed bf16: sT[f,i].
// grid BS, block 128 (thread = output feature f)
// ---------------------------------------------------------------------------
__global__ void k_linear(const float* __restrict__ x, const float* __restrict__ W,
                         bf16_t* __restrict__ sT) {
    __shared__ float xr[FIN];
    const int row = blockIdx.x, f = threadIdx.x;
    xr[f] = x[(size_t)row * FIN + f];
    __syncthreads();
    const float4* w4 = (const float4*)(W + (size_t)f * FIN);
    float acc = 0.f;
    #pragma unroll 8
    for (int i = 0; i < FIN / 4; ++i) {
        float4 wv = w4[i];
        acc += xr[4*i+0]*wv.x + xr[4*i+1]*wv.y + xr[4*i+2]*wv.z + xr[4*i+3]*wv.w;
    }
    sT[(size_t)f * BS + row] = (bf16_t)acc;
}

// ---------------------------------------------------------------------------
// K2: gram tile via WMMA -> d2 -> p = exp(exp(-sqrt(max(d2,1e-12)))).
// Writes raw (unnormalized) p to Praw[BS,BS] bf16 and rowsum[BS].
// grid 256 (16-row strips), block 256 (8 waves; wave w does col tiles w,w+8,..)
// ---------------------------------------------------------------------------
__global__ void k_gram_softmax(const bf16_t* __restrict__ xb, const float* __restrict__ sq,
                               bf16_t* __restrict__ Praw, float* __restrict__ rowsum) {
    const int r0   = blockIdx.x * 16;
    const int wave = threadIdx.x >> 5;
    const int lane = threadIdx.x & 31;
    const int mrow = lane & 15;
    const int half = lane >> 4;

    // A fragments: rows r0..r0+15, K = 0..127 (4 fragments of K=32)
    const bf16_t* arow = xb + (size_t)(r0 + mrow) * FIN;
    v16bf af[4];
    #pragma unroll
    for (int kk = 0; kk < 4; ++kk) af[kk] = ld_afrag(arow + kk * 32 + half * 16);

    float sqi[8];
    {
        const float* sp = sq + r0 + half * 8;
        #pragma unroll
        for (int r = 0; r < 8; ++r) sqi[r] = sp[r];
    }

    float racc[8];
    #pragma unroll
    for (int r = 0; r < 8; ++r) racc[r] = 0.f;

    for (int t = 0; t < 32; ++t) {
        const int jt = wave + (t << 3);       // column tile index (0..255)
        const int n  = jt * 16 + mrow;        // this lane's column
        const bf16_t* brow = xb + (size_t)n * FIN;
        v8f c = {};
        #pragma unroll
        for (int kk = 0; kk < 4; ++kk)
            c = wmma_bf16(af[kk], ld_bfrag(brow + kk * 32 + half * 16), c);

        const float sqn = sq[n];
        bf16_t* pp = Praw + (size_t)(r0 + half * 8) * BS + n;
        #pragma unroll
        for (int r = 0; r < 8; ++r) {
            float d2 = sqi[r] + sqn - 2.f * c[r];
            d2 = fmaxf(d2, 1e-12f);
            float p = __expf(__expf(-sqrtf(d2)));   // exp(dist_exp); inputs in (0,1] -> safe
            racc[r] += p;
            pp[(size_t)r * BS] = (bf16_t)p;
        }
    }

    // reduce row partial sums across the 16 lanes of each half-wave
    __shared__ float partial[8][16];
    float rsum[8];
    #pragma unroll
    for (int r = 0; r < 8; ++r) {
        float v = racc[r];
        #pragma unroll
        for (int m = 8; m >= 1; m >>= 1) v += __shfl_xor(v, m, 16);
        rsum[r] = v;
    }
    if (mrow == 0) {
        #pragma unroll
        for (int r = 0; r < 8; ++r) partial[wave][half * 8 + r] = rsum[r];
    }
    __syncthreads();
    if (threadIdx.x < 16) {
        float s = 0.f;
        #pragma unroll
        for (int w = 0; w < 8; ++w) s += partial[w][threadIdx.x];
        rowsum[r0 + threadIdx.x] = s;
    }
}

// ---------------------------------------------------------------------------
// K3: normalize A = Praw / rowsum (in place) and write Atm = A^T (bf16).
// grid (256 colblocks, 256 rowblocks), block 256 = one 16x16 tile
// ---------------------------------------------------------------------------
__global__ void k_norm_transpose(bf16_t* __restrict__ A, const float* __restrict__ rowsum,
                                 bf16_t* __restrict__ Atm) {
    __shared__ float tile[16][17];
    const int tx = threadIdx.x & 15, ty = threadIdx.x >> 4;
    const int r0 = blockIdx.y * 16, c0 = blockIdx.x * 16;
    const size_t idx = (size_t)(r0 + ty) * BS + (c0 + tx);
    const float v = (float)A[idx] / rowsum[r0 + ty];
    A[idx] = (bf16_t)v;
    tile[ty][tx] = v;
    __syncthreads();
    Atm[(size_t)(c0 + ty) * BS + (r0 + tx)] = (bf16_t)tile[tx][ty];
}

// ---------------------------------------------------------------------------
// K4: invAl[k] = 1 / sum_i A[i,k]  (row sums of Atm).  block 256 = 8 waves,
// wave handles one k; grid 512.
// ---------------------------------------------------------------------------
__global__ void k_colsum(const bf16_t* __restrict__ Atm, float* __restrict__ invAl) {
    const int wave = threadIdx.x >> 5, lane = threadIdx.x & 31;
    const int k = blockIdx.x * 8 + wave;
    const bf16_t* row = Atm + (size_t)k * BS;
    float s = 0.f;
    for (int i = lane; i < BS; i += 32) s += (float)row[i];
    #pragma unroll
    for (int m = 16; m >= 1; m >>= 1) s += __shfl_xor(s, m, 32);
    if (lane == 0) invAl[k] = 1.0f / s;
}

// ---------------------------------------------------------------------------
// K5: t = A^T @ s  (M=4096 rows of Atm, N=128, K=4096), epilogue scales by
// invAl and stores t'^T bf16 (tT[f,k]).  grid 256, block 32 (one wave/M-tile)
// ---------------------------------------------------------------------------
__global__ void k_gemm_t(const bf16_t* __restrict__ Atm, const bf16_t* __restrict__ sT,
                         const float* __restrict__ invAl, bf16_t* __restrict__ tT) {
    const int Mb   = blockIdx.x * 16;
    const int lane = threadIdx.x & 31;
    const int mrow = lane & 15;
    const int half = lane >> 4;

    v8f acc[8];
    #pragma unroll
    for (int nt = 0; nt < 8; ++nt) acc[nt] = (v8f){};

    const bf16_t* arow = Atm + (size_t)(Mb + mrow) * BS;
    for (int k0 = 0; k0 < BS; k0 += 32) {
        __builtin_prefetch(arow + k0 + 512, 0, 1);      // global_prefetch_b8
        v16bf a = ld_afrag(arow + k0 + half * 16);
        #pragma unroll
        for (int nt = 0; nt < 8; ++nt) {
            const bf16_t* bp = sT + (size_t)(nt * 16 + mrow) * BS + k0 + half * 16;
            acc[nt] = wmma_bf16(a, ld_bfrag(bp), acc[nt]);
        }
    }

    const int mb = Mb + half * 8;
    float inv[8];
    #pragma unroll
    for (int r = 0; r < 8; ++r) inv[r] = invAl[mb + r];
    #pragma unroll
    for (int nt = 0; nt < 8; ++nt) {
        const int n = nt * 16 + mrow;
        v8bf o;
        #pragma unroll
        for (int r = 0; r < 8; ++r) o[r] = (bf16_t)(acc[nt][r] * inv[r]);
        *(v8bf*)(tT + (size_t)n * BS + mb) = o;         // contiguous 16B store
    }
}

// ---------------------------------------------------------------------------
// K6: h = leaky_relu(A @ t' + b), f32 out [4096,128]. grid 256, block 32.
// ---------------------------------------------------------------------------
__global__ void k_gemm_h(const bf16_t* __restrict__ A, const bf16_t* __restrict__ tT,
                         const float* __restrict__ bvec, float* __restrict__ out) {
    const int Mb   = blockIdx.x * 16;
    const int lane = threadIdx.x & 31;
    const int mrow = lane & 15;
    const int half = lane >> 4;

    v8f acc[8];
    #pragma unroll
    for (int nt = 0; nt < 8; ++nt) acc[nt] = (v8f){};

    const bf16_t* arow = A + (size_t)(Mb + mrow) * BS;
    for (int k0 = 0; k0 < BS; k0 += 32) {
        __builtin_prefetch(arow + k0 + 512, 0, 1);
        v16bf a = ld_afrag(arow + k0 + half * 16);
        #pragma unroll
        for (int nt = 0; nt < 8; ++nt) {
            const bf16_t* bp = tT + (size_t)(nt * 16 + mrow) * BS + k0 + half * 16;
            acc[nt] = wmma_bf16(a, ld_bfrag(bp), acc[nt]);
        }
    }

    const int mb = Mb + half * 8;
    #pragma unroll
    for (int nt = 0; nt < 8; ++nt) {
        const int n = nt * 16 + mrow;
        const float bn = bvec[n];
        #pragma unroll
        for (int r = 0; r < 8; ++r) {
            float v = acc[nt][r] + bn;
            v = (v >= 0.f) ? v : NEG_SLOPE * v;
            out[(size_t)(mb + r) * FIN + n] = v;
        }
    }
}

// ---------------------------------------------------------------------------
extern "C" void kernel_launch(void* const* d_in, const int* in_sizes, int n_in,
                              void* d_out, int out_size, void* d_ws, size_t ws_size,
                              hipStream_t stream) {
    const float* x = (const float*)d_in[0];   // [4096,128]
    const float* W = (const float*)d_in[1];   // [128,128]
    const float* b = (const float*)d_in[2];   // [128]
    float* out = (float*)d_out;               // [4096,128]

    char* ws = (char*)d_ws;
    size_t off = 0;
    auto carve = [&](size_t bytes) { char* p = ws + off; off += (bytes + 255) & ~(size_t)255; return p; };

    bf16_t* xb    = (bf16_t*)carve((size_t)BS * FIN * 2);   // x in bf16
    float*  sq    = (float*) carve((size_t)BS * 4);         // row norms
    bf16_t* sT    = (bf16_t*)carve((size_t)FIN * BS * 2);   // (x@W.T)^T bf16
    bf16_t* Amat  = (bf16_t*)carve((size_t)BS * BS * 2);    // raw p, then A
    bf16_t* Atm   = (bf16_t*)carve((size_t)BS * BS * 2);    // A^T
    float*  rsum  = (float*) carve((size_t)BS * 4);         // softmax row sums
    float*  invAl = (float*) carve((size_t)BS * 4);         // 1/Alsum
    bf16_t* tT    = (bf16_t*)carve((size_t)FIN * BS * 2);   // (t/Alsum)^T bf16

    k_prep        <<<BS, 32,  0, stream>>>(x, xb, sq);
    k_linear      <<<BS, 128, 0, stream>>>(x, W, sT);
    k_gram_softmax<<<BS / 16, 256, 0, stream>>>(xb, sq, Amat, rsum);
    k_norm_transpose<<<dim3(BS / 16, BS / 16), 256, 0, stream>>>(Amat, rsum, Atm);
    k_colsum      <<<BS / 8, 256, 0, stream>>>(Atm, invAl);
    k_gemm_t      <<<BS / 16, 32, 0, stream>>>(Atm, sT, invAl, tT);
    k_gemm_h      <<<BS / 16, 32, 0, stream>>>(Amat, tT, b, out);
}
